// Block_19129784336906
// MI455X (gfx1250) — matrix operand order repaired and checked by reading, not compiled, not run on hardware
//
#include <hip/hip_runtime.h>
#include <cstdint>
#include <cstddef>

typedef __attribute__((ext_vector_type(16))) _Float16 v16h;
typedef __attribute__((ext_vector_type(8)))  float    v8f;

#define HEADS    16
#define HEAD_DIM 64
#define HIDDEN   1024
#define QKVFF    7168
#define VFF      5120
#define BATCH    2
#define SEQ      2048
#define NTOK     (BATCH * SEQ)
#define LN_EPS   1e-5f

// ---------------------------------------------------------------------------
// WMMA fragment loaders (CDNA5 ISA 7.12.2 layouts, wave32)
// A-matrix 16x32 f16: lane holds row M=lane%16; K interleaved by 8 per half:
//   dwords 0-3 = row dwords [4*half .. 4*half+3], dwords 4-7 = [8+4*half ..]
// B-matrix 32x16 f16: lane holds column N=lane%16; K interleaved by 8 per half
//   (same as A: dwords 0-3 = [4*half ..], dwords 4-7 = [8+4*half ..])
// ---------------------------------------------------------------------------
union FragU {
    v16h h;
    int4 q[2];
};

static __device__ __forceinline__ v16h lds_frag_a(const _Float16* base, int strideH,
                                                  int row0, int col0, int lane) {
    const int r = row0 + (lane & 15);
    const int hs = lane >> 4;
    const int4* p = (const int4*)(base + (size_t)r * strideH + col0);
    FragU f;
    f.q[0] = p[hs];
    f.q[1] = p[2 + hs];
    return f.h;
}

static __device__ __forceinline__ v16h lds_frag_b(const _Float16* base, int strideH,
                                                  int row0, int col0, int lane) {
    const int r = row0 + (lane & 15);
    const int hs = lane >> 4;
    const int4* p = (const int4*)(base + (size_t)r * strideH + col0);
    FragU f;
    f.q[0] = p[hs];
    f.q[1] = p[2 + hs];
    return f.h;
}

// ---------------------------------------------------------------------------
// K0: fp32 -> f16 cast (weights)
// ---------------------------------------------------------------------------
__global__ void cast_f32_f16(const float* __restrict__ src, _Float16* __restrict__ dst, int n) {
    int i = blockIdx.x * blockDim.x + threadIdx.x;
    int stride = gridDim.x * blockDim.x;
    for (; i < n; i += stride) dst[i] = (_Float16)src[i];
}

// ---------------------------------------------------------------------------
// K1: LayerNorm per token (256 threads / row), f16 out
// ---------------------------------------------------------------------------
__global__ __launch_bounds__(256) void layernorm_kernel(
    const float* __restrict__ x, const float* __restrict__ w,
    const float* __restrict__ b, _Float16* __restrict__ out)
{
    const int t = blockIdx.x;
    const float* row = x + (size_t)t * HIDDEN;
    float v0[4];
    float s = 0.f, ss = 0.f;
#pragma unroll
    for (int i = 0; i < 4; ++i) {
        float v = row[threadIdx.x + 256 * i];
        v0[i] = v; s += v; ss += v * v;
    }
#pragma unroll
    for (int off = 16; off >= 1; off >>= 1) {
        s  += __shfl_xor(s,  off, 32);
        ss += __shfl_xor(ss, off, 32);
    }
    __shared__ float ps[8], pss[8];
    const int wave = threadIdx.x >> 5, lane = threadIdx.x & 31;
    if (lane == 0) { ps[wave] = s; pss[wave] = ss; }
    __syncthreads();
    if (wave == 0) {
        float a  = (lane < 8) ? ps[lane]  : 0.f;
        float a2 = (lane < 8) ? pss[lane] : 0.f;
#pragma unroll
        for (int off = 4; off >= 1; off >>= 1) {
            a  += __shfl_xor(a,  off, 32);
            a2 += __shfl_xor(a2, off, 32);
        }
        if (lane == 0) { ps[0] = a; pss[0] = a2; }
    }
    __syncthreads();
    const float mean = ps[0] * (1.0f / HIDDEN);
    const float var  = pss[0] * (1.0f / HIDDEN) - mean * mean;
    const float rstd = rsqrtf(var + LN_EPS);
    _Float16* orow = out + (size_t)t * HIDDEN;
#pragma unroll
    for (int i = 0; i < 4; ++i) {
        int c = threadIdx.x + 256 * i;
        orow[c] = (_Float16)((v0[i] - mean) * rstd * w[c] + b[c]);
    }
}

// ---------------------------------------------------------------------------
// K2/K5: f16 GEMM, C(f32)[M,N] = A(f16)[M,K] * B(f16)[K,N]
// Compile-time N,K. 128x128x64 macrotile, 8 waves (4x2), 16 WMMAs per stage.
// Double-buffered LDS: one barrier per k-stage; next tile's global loads are
// issued before computing the current tile. global_prefetch_b8 covers k+2.
// ---------------------------------------------------------------------------
#define BM 128
#define BN 128
#define BK 64
#define ASTR 72   // halfs; 144B rows, 16B aligned
#define BSTR 72

union Row8 { int4 v; _Float16 e[8]; };

template <int N, int K>
__global__ __launch_bounds__(256) void gemm_f16_wmma(
    const _Float16* __restrict__ A, const _Float16* __restrict__ B,
    float* __restrict__ C)
{
    __shared__ _Float16 As[2][BM * ASTR];   // [m][k]
    __shared__ _Float16 Bs[2][BN * BSTR];   // transposed: [n][k]

    const int tid  = threadIdx.x;
    const int lane = tid & 31;
    const int wave = tid >> 5;
    const int wm   = wave & 3;   // row group of 32
    const int wn   = wave >> 2;  // col group of 64
    const int n0   = blockIdx.x * BN;
    const int m0   = blockIdx.y * BM;

    v8f vzero = {};
    v8f acc[2][4];
#pragma unroll
    for (int i = 0; i < 2; ++i)
#pragma unroll
        for (int j = 0; j < 4; ++j) acc[i][j] = vzero;

    const int ar = tid >> 1;   // 0..127  A row
    const int ap = tid & 1;    // 32-half chunk of the row
    const int bq = tid >> 4;   // 0..15 -> k base bq*4
    const int bn = tid & 15;   // 8-wide n chunk

    const _Float16* aptr = A + (size_t)(m0 + ar) * K + ap * 32;
    const _Float16* bptr = B + (size_t)(bq * 4) * N + n0 + bn * 8;

    int4 areg[4];
    Row8 breg[4];

    auto load_tile = [&](int k0) {
        const int4* asrc = (const int4*)(aptr + k0);
#pragma unroll
        for (int i = 0; i < 4; ++i) areg[i] = asrc[i];
#pragma unroll
        for (int j = 0; j < 4; ++j)
            breg[j].v = *(const int4*)(bptr + (size_t)(k0 + j) * N);
    };
    auto store_tile = [&](int buf) {
        int4* adst = (int4*)(As[buf] + (size_t)ar * ASTR + ap * 32);
#pragma unroll
        for (int i = 0; i < 4; ++i) adst[i] = areg[i];
#pragma unroll
        for (int i = 0; i < 8; ++i) {
            union { _Float16 e[4]; int2 v; } pk;
            pk.e[0] = breg[0].e[i];
            pk.e[1] = breg[1].e[i];
            pk.e[2] = breg[2].e[i];
            pk.e[3] = breg[3].e[i];
            *(int2*)(Bs[buf] + (size_t)(bn * 8 + i) * BSTR + bq * 4) = pk.v;
        }
    };
    auto compute = [&](int buf) {
#pragma unroll
        for (int ks = 0; ks < 2; ++ks) {
            v16h af[2], bf[4];
#pragma unroll
            for (int i = 0; i < 2; ++i)
                af[i] = lds_frag_a(As[buf], ASTR, wm * 32 + i * 16, ks * 32, lane);
#pragma unroll
            for (int j = 0; j < 4; ++j)
                bf[j] = lds_frag_b(Bs[buf], BSTR, wn * 64 + j * 16, ks * 32, lane);
#pragma unroll
            for (int i = 0; i < 2; ++i)
#pragma unroll
                for (int j = 0; j < 4; ++j)
                    acc[i][j] = __builtin_amdgcn_wmma_f32_16x16x32_f16(
                        false, af[i], false, bf[j], (short)0, acc[i][j], false, false);
        }
    };

    constexpr int T = K / BK;
    load_tile(0);
    store_tile(0);
    __syncthreads();
    for (int kt = 0; kt < T; ++kt) {
        const int cur = kt & 1;
        if (kt + 1 < T) {
            load_tile((kt + 1) * BK);               // hide global latency behind WMMAs
            if (kt + 2 < T) {
                __builtin_prefetch(aptr + (kt + 2) * BK, 0, 1);
                __builtin_prefetch(bptr + (size_t)(kt + 2) * BK * N, 0, 1);
            }
        }
        compute(cur);
        if (kt + 1 < T) store_tile(cur ^ 1);        // other buffer: no race with readers
        __syncthreads();
    }

    // Epilogue: one base pointer per lane; all offsets are compile-time.
    const int hs = lane >> 4;
    const int ln = lane & 15;
    float* cbase = C + (size_t)(m0 + wm * 32 + 8 * hs) * N + n0 + wn * 64 + ln;
#pragma unroll
    for (int i = 0; i < 2; ++i)
#pragma unroll
        for (int j = 0; j < 4; ++j)
#pragma unroll
            for (int jv = 0; jv < 8; ++jv)
                cbase[(size_t)(i * 16 + jv) * N + j * 16] = acc[i][j][jv];
}

// ---------------------------------------------------------------------------
// K3: RoPE(q,k) + reorder to [B,H,L,D] f16, V copy, GELU(ff) -> concat cols
// ---------------------------------------------------------------------------
__global__ __launch_bounds__(256) void rope_split_kernel(
    const float* __restrict__ h,
    _Float16* __restrict__ Q, _Float16* __restrict__ Kc,
    _Float16* __restrict__ V, _Float16* __restrict__ concat)
{
    const int t = blockIdx.x;
    const int b = t / SEQ;
    const int l = t % SEQ;
    const float* hr = h + (size_t)t * QKVFF;
    const float lpos = (float)l;

    for (int e = threadIdx.x; e < HIDDEN; e += 256) {
        const int head = e >> 6;
        const int d    = e & 63;
        const int f    = d & 31;
        const float inv = __expf(-(float)(2 * f) * (1.0f / HEAD_DIM) * 9.210340371976184f); // ln(1e4)
        float sn, cs;
        __sincosf(lpos * inv, &sn, &cs);
        const size_t oidx = (((size_t)b * HEADS + head) * SEQ + l) * HEAD_DIM + d;
        const int partner = (d < 32) ? e + 32 : e - 32;
        {
            float xq = hr[e], xp = hr[partner];
            Q[oidx] = (_Float16)((d < 32) ? (xq * cs - xp * sn) : (xq * cs + xp * sn));
        }
        {
            float xk = hr[HIDDEN + e], xp = hr[HIDDEN + partner];
            Kc[oidx] = (_Float16)((d < 32) ? (xk * cs - xp * sn) : (xk * cs + xp * sn));
        }
        V[oidx] = (_Float16)hr[2 * HIDDEN + e];
    }

    _Float16* crow = concat + (size_t)t * VFF + HIDDEN;
    const float* ffr = hr + 3 * HIDDEN;
    for (int e = threadIdx.x; e < 4 * HIDDEN; e += 256) {
        float xv = ffr[e];
        crow[e] = (_Float16)(0.5f * xv * (1.0f + erff(xv * 0.70710678118654752f)));
    }
}

// ---------------------------------------------------------------------------
// K4: causal flash attention, 64 q-rows per block, 4 waves x 16 rows
// ---------------------------------------------------------------------------
#define ATS 72   // LDS row stride in halfs for 64-wide tiles (144B, 16B-aligned)

__global__ __launch_bounds__(128) void attn_kernel(
    const _Float16* __restrict__ Q, const _Float16* __restrict__ Kc,
    const _Float16* __restrict__ V, _Float16* __restrict__ concat)
{
    __shared__ _Float16 Qs[64 * ATS];
    __shared__ _Float16 Ks[64 * ATS];
    __shared__ _Float16 Vt[64 * ATS];           // transposed: [d][key]
    __shared__ _Float16 Ps[4][16 * ATS];        // per-wave P staging

    const int tid  = threadIdx.x;
    const int lane = tid & 31;
    const int wave = tid >> 5;
    const int qblk = blockIdx.x;   // 0..31
    const int hh   = blockIdx.y;   // head
    const int bb   = blockIdx.z;   // batch

    const size_t hbase = ((size_t)bb * HEADS + hh) * SEQ * HEAD_DIM;
    const int q0 = qblk * 64;

    {   // stage Q tile [64][64]
        const int r = tid >> 1, p = tid & 1;
        const int4* src = (const int4*)(Q + hbase + (size_t)(q0 + r) * HEAD_DIM + p * 32);
        int4* dst = (int4*)(Qs + (size_t)r * ATS + p * 32);
        dst[0] = src[0]; dst[1] = src[1]; dst[2] = src[2]; dst[3] = src[3];
    }
    __syncthreads();
    // Q fragments are loop-invariant: load once.
    v16h aq0 = lds_frag_a(Qs, ATS, wave * 16, 0,  lane);
    v16h aq1 = lds_frag_a(Qs, ATS, wave * 16, 32, lane);

    v8f vzero = {};
    v8f acco[4];
#pragma unroll
    for (int n = 0; n < 4; ++n) acco[n] = vzero;
    float mrow[8], lrow[8];
#pragma unroll
    for (int jv = 0; jv < 8; ++jv) { mrow[jv] = -1e30f; lrow[jv] = 0.f; }

    const int hs = lane >> 4;
    const int ln = lane & 15;
    const float scale = 0.125f;  // 1/sqrt(64)

    for (int jb = 0; jb <= qblk; ++jb) {
        __syncthreads();
        {   // stage K tile [key][d]
            const int r = tid >> 1, p = tid & 1;
            const int4* src = (const int4*)(Kc + hbase + (size_t)(jb * 64 + r) * HEAD_DIM + p * 32);
            int4* dst = (int4*)(Ks + (size_t)r * ATS + p * 32);
            dst[0] = src[0]; dst[1] = src[1]; dst[2] = src[2]; dst[3] = src[3];
        }
        {   // stage V transposed -> Vt[d][key], packed 4 keys -> ds_store_b64
            const int rq = tid >> 3;   // 0..15 -> key rows rq*4..+3
            const int dp = tid & 7;    // 8-wide d chunk
            Row8 vb[4];
#pragma unroll
            for (int j = 0; j < 4; ++j)
                vb[j].v = *(const int4*)(V + hbase + (size_t)(jb * 64 + rq * 4 + j) * HEAD_DIM + dp * 8);
#pragma unroll
            for (int i = 0; i < 8; ++i) {
                union { _Float16 e[4]; int2 v; } pk;
                pk.e[0] = vb[0].e[i];
                pk.e[1] = vb[1].e[i];
                pk.e[2] = vb[2].e[i];
                pk.e[3] = vb[3].e[i];
                *(int2*)(Vt + (size_t)(dp * 8 + i) * ATS + rq * 4) = pk.v;
            }
        }
        __syncthreads();

        // S = Q x K^T for this wave's 16 q rows; preload all B fragments
        v16h bk[4][2];
#pragma unroll
        for (int n = 0; n < 4; ++n) {
            bk[n][0] = lds_frag_b(Ks, ATS, n * 16, 0,  lane);
            bk[n][1] = lds_frag_b(Ks, ATS, n * 16, 32, lane);
        }
        v8f s[4];
#pragma unroll
        for (int n = 0; n < 4; ++n) s[n] = vzero;
#pragma unroll
        for (int n = 0; n < 4; ++n) {
            s[n] = __builtin_amdgcn_wmma_f32_16x16x32_f16(false, aq0, false, bk[n][0], (short)0, s[n], false, false);
            s[n] = __builtin_amdgcn_wmma_f32_16x16x32_f16(false, aq1, false, bk[n][1], (short)0, s[n], false, false);
        }

        // scale + causal mask + online softmax (row stats across 16-lane groups)
        const int qg_base = q0 + wave * 16 + 8 * hs;
        const int kg_base = jb * 64 + ln;
        float pbuf[4][8];
#pragma unroll
        for (int jv = 0; jv < 8; ++jv) {
            const int qg = qg_base + jv;
            float mloc = -1e30f;
#pragma unroll
            for (int n = 0; n < 4; ++n) {
                float val = s[n][jv] * scale;
                if (kg_base + n * 16 > qg) val = -1e30f;
                pbuf[n][jv] = val;
                mloc = fmaxf(mloc, val);
            }
#pragma unroll
            for (int off = 1; off < 16; off <<= 1)
                mloc = fmaxf(mloc, __shfl_xor(mloc, off, 32));
            const float mnew  = fmaxf(mrow[jv], mloc);
            const float alpha = __expf(mrow[jv] - mnew);
            float rowsum = 0.f;
#pragma unroll
            for (int n = 0; n < 4; ++n) {
                float p = __expf(pbuf[n][jv] - mnew);
                pbuf[n][jv] = p;
                rowsum += p;
            }
#pragma unroll
            for (int off = 1; off < 16; off <<= 1)
                rowsum += __shfl_xor(rowsum, off, 32);
            lrow[jv] = lrow[jv] * alpha + rowsum;
            mrow[jv] = mnew;
#pragma unroll
            for (int n = 0; n < 4; ++n)
                acco[n][jv] = acco[n][jv] * alpha;
        }

        // stash P (f16) in per-wave LDS region, layout [row][key]
#pragma unroll
        for (int jv = 0; jv < 8; ++jv) {
            const int rloc = jv + 8 * hs;
#pragma unroll
            for (int n = 0; n < 4; ++n)
                Ps[wave][(size_t)rloc * ATS + n * 16 + ln] = (_Float16)pbuf[n][jv];
        }

        // O += P x V; preload all fragments
        v16h ap0 = lds_frag_a(Ps[wave], ATS, 0, 0,  lane);
        v16h ap1 = lds_frag_a(Ps[wave], ATS, 0, 32, lane);
        v16h bv[4][2];
#pragma unroll
        for (int n = 0; n < 4; ++n) {
            bv[n][0] = lds_frag_b(Vt, ATS, n * 16, 0,  lane);
            bv[n][1] = lds_frag_b(Vt, ATS, n * 16, 32, lane);
        }
#pragma unroll
        for (int n = 0; n < 4; ++n) {
            acco[n] = __builtin_amdgcn_wmma_f32_16x16x32_f16(false, ap0, false, bv[n][0], (short)0, acco[n], false, false);
            acco[n] = __builtin_amdgcn_wmma_f32_16x16x32_f16(false, ap1, false, bv[n][1], (short)0, acco[n], false, false);
        }
    }

    // finalize: O /= l ; write to concat[:, h*64 + d]
#pragma unroll
    for (int jv = 0; jv < 8; ++jv) {
        const int qg = q0 + wave * 16 + 8 * hs + jv;
        const size_t tok = (size_t)bb * SEQ + qg;
        const float invl = 1.0f / lrow[jv];
#pragma unroll
        for (int n = 0; n < 4; ++n) {
            const int d = n * 16 + ln;
            concat[tok * VFF + hh * HEAD_DIM + d] = (_Float16)(acco[n][jv] * invl);
        }
    }
}

// ---------------------------------------------------------------------------
// Host-side launch
// ---------------------------------------------------------------------------
extern "C" void kernel_launch(void* const* d_in, const int* in_sizes, int n_in,
                              void* d_out, int out_size, void* d_ws, size_t ws_size,
                              hipStream_t stream)
{
    (void)in_sizes; (void)n_in; (void)out_size; (void)ws_size;
    const float* x     = (const float*)d_in[0];
    const float* ln_w  = (const float*)d_in[1];
    const float* ln_b  = (const float*)d_in[2];
    const float* W_in  = (const float*)d_in[3];
    const float* W_out = (const float*)d_in[4];
    float* out = (float*)d_out;

    char* ws = (char*)d_ws;
    size_t off = 0;
    auto carve = [&](size_t bytes) -> void* {
        void* p = ws + off;
        off += (bytes + 255) & ~(size_t)255;
        return p;
    };
    _Float16* Win_h  = (_Float16*)carve((size_t)HIDDEN * QKVFF * 2);
    _Float16* Wout_h = (_Float16*)carve((size_t)VFF * HIDDEN * 2);
    _Float16* xn_h   = (_Float16*)carve((size_t)NTOK * HIDDEN * 2);
    float*    hbuf   = (float*)   carve((size_t)NTOK * QKVFF * 4);
    _Float16* Qb     = (_Float16*)carve((size_t)NTOK * HIDDEN * 2);
    _Float16* Kb     = (_Float16*)carve((size_t)NTOK * HIDDEN * 2);
    _Float16* Vb     = (_Float16*)carve((size_t)NTOK * HIDDEN * 2);
    _Float16* Cc     = (_Float16*)carve((size_t)NTOK * VFF * 2);

    cast_f32_f16<<<2048, 256, 0, stream>>>(W_in,  Win_h,  HIDDEN * QKVFF);
    cast_f32_f16<<<2048, 256, 0, stream>>>(W_out, Wout_h, VFF * HIDDEN);
    layernorm_kernel<<<NTOK, 256, 0, stream>>>(x, ln_w, ln_b, xn_h);
    gemm_f16_wmma<QKVFF, HIDDEN><<<dim3(QKVFF / BN, NTOK / BM), 256, 0, stream>>>(
        xn_h, Win_h, hbuf);
    rope_split_kernel<<<NTOK, 256, 0, stream>>>(hbuf, Qb, Kb, Vb, Cc);
    attn_kernel<<<dim3(SEQ / 64, HEADS, BATCH), 128, 0, stream>>>(Qb, Kb, Vb, Cc);
    gemm_f16_wmma<HIDDEN, VFF><<<dim3(HIDDEN / BN, NTOK / BM), 256, 0, stream>>>(
        Cc, Wout_h, out);
}
